// GCN_77180562309590
// MI455X (gfx1250) — matrix operand order repaired and checked by reading, not compiled, not run on hardware
//
#include <hip/hip_runtime.h>
#include <hip/hip_bf16.h>

// ---------------------------------------------------------------------------
// GCN forward for MI455X (gfx1250): f16 WMMA GEMMs + atomic message passing.
// ---------------------------------------------------------------------------

static constexpr int N_  = 50000;
static constexpr int E_  = 800000;
static constexpr int IN_ = 128;
static constexpr int D_  = 64;
static constexpr int G_  = 512;
static constexpr float EPS_   = 1e-5f;
static constexpr float SLOPE_ = 0.01f;

typedef __attribute__((ext_vector_type(16))) _Float16 v16h;
typedef __attribute__((ext_vector_type(8)))  _Float16 v8h;
typedef __attribute__((ext_vector_type(8)))  float    v8f;

union Frag16 { v16h v; v8h h2[2]; _Float16 e[16]; };

// ---------------------------------------------------------------------------
// Degree / normalization
// ---------------------------------------------------------------------------
__global__ __launch_bounds__(256) void k_deg_init(float* __restrict__ deg, int n) {
    int i = blockIdx.x * blockDim.x + threadIdx.x;
    if (i < n) deg[i] = 1.0f;                       // self-loop weight 1
}

__global__ __launch_bounds__(256) void k_deg_accum(const int* __restrict__ dst,
                                                   const float* __restrict__ w,
                                                   float* __restrict__ deg, int e) {
    int i = blockIdx.x * blockDim.x + threadIdx.x;
    if (i < e) atomicAdd(&deg[dst[i]], w[i]);
}

__global__ __launch_bounds__(256) void k_dinv(float* __restrict__ deg,      // in: deg, out: dinv
                                              float* __restrict__ selfn, int n) {
    int i = blockIdx.x * blockDim.x + threadIdx.x;
    if (i >= n) return;
    float dg = deg[i];
    float dv = dg > 0.0f ? rsqrtf(dg) : 0.0f;
    deg[i]   = dv;
    selfn[i] = dv * dv;                              // self-loop norm (weight 1)
}

__global__ __launch_bounds__(256) void k_edge_norm(const int* __restrict__ src,
                                                   const int* __restrict__ dst,
                                                   const float* __restrict__ w,
                                                   const float* __restrict__ dinv,
                                                   float* __restrict__ norm, int e) {
    int i = blockIdx.x * blockDim.x + threadIdx.x;
    if (i < e) norm[i] = dinv[src[i]] * w[i] * dinv[dst[i]];
}

// ---------------------------------------------------------------------------
// WMMA GEMM:  out[N,64] = A[N,K] * W[K,64]  (f32 in/out, f16 matrix pipes)
// One wave -> 16x64 output strip; 4 col-tiles, K-loop of K/32 WMMA steps.
// ---------------------------------------------------------------------------
template<int K>
__global__ __launch_bounds__(128)
void k_gemm_wmma(const float* __restrict__ A, const float* __restrict__ W,
                 float* __restrict__ out, int nRowTiles) {
    __shared__ _Float16 Wt[64 * K];                  // transposed: Wt[col*K + k]
    for (int idx = threadIdx.x; idx < K * 64; idx += 128) {
        const int k = idx >> 6, c = idx & 63;
        Wt[c * K + k] = (_Float16)W[idx];
    }
    __syncthreads();

    const int lane = threadIdx.x & 31;
    const int tile = blockIdx.x * 4 + (threadIdx.x >> 5);
    if (tile >= nRowTiles) return;

    const int m  = lane & 15;
    const int hi = lane >> 4;
    const float* __restrict__ arow = A + (size_t)(tile * 16 + m) * K;

    v8f acc0 = {}, acc1 = {}, acc2 = {}, acc3 = {};

#pragma unroll
    for (int k0 = 0; k0 < K; k0 += 32) {
        // ---- A fragment (16x32 f16): e0..7 -> k0+8*hi+0..7, e8..15 -> k0+16+8*hi+0..7
        Frag16 a;
        {
            const float4 p0 = *(const float4*)&arow[k0 + hi * 8];
            const float4 p1 = *(const float4*)&arow[k0 + hi * 8 + 4];
            const float4 p2 = *(const float4*)&arow[k0 + 16 + hi * 8];
            const float4 p3 = *(const float4*)&arow[k0 + 16 + hi * 8 + 4];
            a.e[0]  = (_Float16)p0.x; a.e[1]  = (_Float16)p0.y;
            a.e[2]  = (_Float16)p0.z; a.e[3]  = (_Float16)p0.w;
            a.e[4]  = (_Float16)p1.x; a.e[5]  = (_Float16)p1.y;
            a.e[6]  = (_Float16)p1.z; a.e[7]  = (_Float16)p1.w;
            a.e[8]  = (_Float16)p2.x; a.e[9]  = (_Float16)p2.y;
            a.e[10] = (_Float16)p2.z; a.e[11] = (_Float16)p2.w;
            a.e[12] = (_Float16)p3.x; a.e[13] = (_Float16)p3.y;
            a.e[14] = (_Float16)p3.z; a.e[15] = (_Float16)p3.w;
        }
        // ---- B fragments (32x16 f16): lane col = m, e0..7 -> k0+8*hi+e, e8..15 -> k0+16+8*hi+(e-8)
        const int kb = k0 + hi * 8;
        Frag16 b0, b1, b2, b3;
        b0.h2[0] = *(const v8h*)&Wt[(0  + m) * K + kb];
        b0.h2[1] = *(const v8h*)&Wt[(0  + m) * K + kb + 16];
        b1.h2[0] = *(const v8h*)&Wt[(16 + m) * K + kb];
        b1.h2[1] = *(const v8h*)&Wt[(16 + m) * K + kb + 16];
        b2.h2[0] = *(const v8h*)&Wt[(32 + m) * K + kb];
        b2.h2[1] = *(const v8h*)&Wt[(32 + m) * K + kb + 16];
        b3.h2[0] = *(const v8h*)&Wt[(48 + m) * K + kb];
        b3.h2[1] = *(const v8h*)&Wt[(48 + m) * K + kb + 16];

        acc0 = __builtin_amdgcn_wmma_f32_16x16x32_f16(false, a.v, false, b0.v, (short)0, acc0, false, false);
        acc1 = __builtin_amdgcn_wmma_f32_16x16x32_f16(false, a.v, false, b1.v, (short)0, acc1, false, false);
        acc2 = __builtin_amdgcn_wmma_f32_16x16x32_f16(false, a.v, false, b2.v, (short)0, acc2, false, false);
        acc3 = __builtin_amdgcn_wmma_f32_16x16x32_f16(false, a.v, false, b3.v, (short)0, acc3, false, false);
    }

    // ---- C/D layout: VGPR r -> row = r + 8*hi, col = m
    float* __restrict__ obase = out + (size_t)(tile * 16 + hi * 8) * 64 + m;
#pragma unroll
    for (int r = 0; r < 8; ++r) {
        float* op = obase + (size_t)r * 64;
        op[0]  = acc0[r];
        op[16] = acc1[r];
        op[32] = acc2[r];
        op[48] = acc3[r];
    }
}

// ---------------------------------------------------------------------------
// Aggregation: out = bias + selfnorm*hw  (init), then edge atomics.
// ---------------------------------------------------------------------------
__global__ __launch_bounds__(256)
void k_agg_init(const float* __restrict__ hw, const float* __restrict__ selfn,
                const float* __restrict__ bias, float* __restrict__ out, int n) {
    int idx = blockIdx.x * blockDim.x + threadIdx.x;          // n*16 threads
    if (idx >= n * 16) return;
    const int i = idx >> 4, c = idx & 15;
    float4 v = ((const float4*)(hw + (size_t)i * 64))[c];
    float4 b = ((const float4*)bias)[c];
    float  s = selfn[i];
    float4 r;
    r.x = b.x + v.x * s; r.y = b.y + v.y * s;
    r.z = b.z + v.z * s; r.w = b.w + v.w * s;
    ((float4*)(out + (size_t)i * 64))[c] = r;
}

__global__ __launch_bounds__(256)
void k_edge_scatter(const float* __restrict__ hw, const int* __restrict__ src,
                    const int* __restrict__ dst, const float* __restrict__ norm,
                    float* __restrict__ out, int e) {
    long long idx = (long long)blockIdx.x * blockDim.x + threadIdx.x;  // e*16 threads
    if (idx >= (long long)e * 16) return;
    const int ei = (int)(idx >> 4), c = (int)(idx & 15);
    const int s = src[ei], d = dst[ei];
    const float nr = norm[ei];
    float4 v = ((const float4*)(hw + (size_t)s * 64))[c];
    float* o = out + (size_t)d * 64 + c * 4;
    atomicAdd(o + 0, v.x * nr);
    atomicAdd(o + 1, v.y * nr);
    atomicAdd(o + 2, v.z * nr);
    atomicAdd(o + 3, v.w * nr);
}

// ---------------------------------------------------------------------------
// BatchNorm (training-mode batch stats) + LeakyReLU
// ---------------------------------------------------------------------------
__global__ __launch_bounds__(256) void k_zero(float* __restrict__ p, int n) {
    int i = blockIdx.x * blockDim.x + threadIdx.x;
    if (i < n) p[i] = 0.0f;
}

__global__ __launch_bounds__(256)
void k_bn_partial(const float* __restrict__ h, float* __restrict__ sum,
                  float* __restrict__ sumsq, int n) {
    __shared__ float sS[256], sQ[256];
    const int f = threadIdx.x & 63;
    const int g = threadIdx.x >> 6;                           // 0..3 row groups
    float s = 0.0f, q = 0.0f;
    const int stride = gridDim.x * 4;
    for (int r = blockIdx.x * 4 + g; r < n; r += stride) {
        float v = h[(size_t)r * 64 + f];
        s += v; q += v * v;
    }
    sS[threadIdx.x] = s; sQ[threadIdx.x] = q;
    __syncthreads();
    if (threadIdx.x < 64) {
        s = sS[threadIdx.x] + sS[threadIdx.x + 64] + sS[threadIdx.x + 128] + sS[threadIdx.x + 192];
        q = sQ[threadIdx.x] + sQ[threadIdx.x + 64] + sQ[threadIdx.x + 128] + sQ[threadIdx.x + 192];
        atomicAdd(&sum[f], s);
        atomicAdd(&sumsq[f], q);
    }
}

__global__ __launch_bounds__(64)
void k_bn_finalize(const float* __restrict__ sum, const float* __restrict__ sumsq,
                   const float* __restrict__ gamma, const float* __restrict__ beta,
                   float* __restrict__ scale, float* __restrict__ shift, int n) {
    int f = threadIdx.x;
    if (f >= 64) return;
    float mean = sum[f] / (float)n;
    float var  = sumsq[f] / (float)n - mean * mean;           // ddof=0 like jnp.var
    float sc   = gamma[f] * rsqrtf(var + EPS_);
    scale[f] = sc;
    shift[f] = beta[f] - mean * sc;
}

__device__ __forceinline__ float leaky1(float x) { return x >= 0.0f ? x : SLOPE_ * x; }

__global__ __launch_bounds__(256)
void k_bn_leaky(float* __restrict__ h, const float* __restrict__ scale,
                const float* __restrict__ shift, int n4) {
    int idx = blockIdx.x * blockDim.x + threadIdx.x;          // n*16 float4s
    if (idx >= n4) return;
    const int f = (idx & 15) * 4;
    float4 v = ((float4*)h)[idx];
    float4 sc = *(const float4*)&scale[f];
    float4 sh = *(const float4*)&shift[f];
    v.x = leaky1(v.x * sc.x + sh.x);
    v.y = leaky1(v.y * sc.y + sh.y);
    v.z = leaky1(v.z * sc.z + sh.z);
    v.w = leaky1(v.w * sc.w + sh.w);
    ((float4*)h)[idx] = v;
}

__global__ __launch_bounds__(256)
void k_leaky(float* __restrict__ h, int n4) {
    int idx = blockIdx.x * blockDim.x + threadIdx.x;
    if (idx >= n4) return;
    float4 v = ((float4*)h)[idx];
    v.x = leaky1(v.x); v.y = leaky1(v.y); v.z = leaky1(v.z); v.w = leaky1(v.w);
    ((float4*)h)[idx] = v;
}

// ---------------------------------------------------------------------------
// Per-graph mean/max pooling + readout
// ---------------------------------------------------------------------------
__global__ __launch_bounds__(256)
void k_pool_init(float* __restrict__ meanacc, float* __restrict__ mx,
                 float* __restrict__ cnt) {
    int idx = blockIdx.x * blockDim.x + threadIdx.x;          // G*64 threads
    if (idx >= G_ * 64) return;
    meanacc[idx] = 0.0f;
    mx[idx]      = -3.402823466e38f;
    if (idx < G_) cnt[idx] = 0.0f;
}

__global__ __launch_bounds__(256)
void k_pool_scatter(const float* __restrict__ h, const int* __restrict__ batch,
                    float* __restrict__ meanacc, float* __restrict__ mx,
                    float* __restrict__ cnt, int n) {
    long long idx = (long long)blockIdx.x * blockDim.x + threadIdx.x;  // n*64
    if (idx >= (long long)n * 64) return;
    const int i = (int)(idx >> 6), f = (int)(idx & 63);
    const int g = batch[i];
    const float v = h[idx];
    atomicAdd(&meanacc[(size_t)g * 64 + f], v);
    __hip_atomic_fetch_max(&mx[(size_t)g * 64 + f], v,
                           __ATOMIC_RELAXED, __HIP_MEMORY_SCOPE_AGENT);
    if (f == 0) atomicAdd(&cnt[g], 1.0f);
}

__global__ __launch_bounds__(256)
void k_readout(const float* __restrict__ meanacc, const float* __restrict__ mx,
               const float* __restrict__ cnt, const float* __restrict__ Wout,
               const float* __restrict__ bout, float* __restrict__ out) {
    int g = blockIdx.x * blockDim.x + threadIdx.x;
    if (g >= G_) return;
    const float c   = cnt[g];
    const float inv = 1.0f / fmaxf(c, 1.0f);
    float acc = bout[0];
#pragma unroll 4
    for (int f = 0; f < 64; ++f) {
        float mval = (c > 0.0f) ? mx[(size_t)g * 64 + f] : 0.0f;   // pooled = [max, mean]
        acc += mval * Wout[f];
        acc += (meanacc[(size_t)g * 64 + f] * inv) * Wout[64 + f];
    }
    out[g] = acc;
}

// ---------------------------------------------------------------------------
// Launch
// ---------------------------------------------------------------------------
extern "C" void kernel_launch(void* const* d_in, const int* in_sizes, int n_in,
                              void* d_out, int out_size, void* d_ws, size_t ws_size,
                              hipStream_t stream) {
    (void)in_sizes; (void)n_in; (void)out_size; (void)ws_size;

    const float* x     = (const float*)d_in[0];
    const int*   eidx  = (const int*)d_in[1];
    const float* eattr = (const float*)d_in[2];
    const int*   batch = (const int*)d_in[3];
    const float* W0 = (const float*)d_in[4];  const float* b0  = (const float*)d_in[5];
    const float* g0 = (const float*)d_in[6];  const float* be0 = (const float*)d_in[7];
    const float* W1 = (const float*)d_in[8];  const float* b1  = (const float*)d_in[9];
    const float* g1 = (const float*)d_in[10]; const float* be1 = (const float*)d_in[11];
    const float* W2 = (const float*)d_in[12]; const float* b2  = (const float*)d_in[13];
    const float* W3 = (const float*)d_in[14]; const float* b3  = (const float*)d_in[15];
    const float* Wout = (const float*)d_in[16]; const float* bout = (const float*)d_in[17];
    float* out = (float*)d_out;

    const int* src = eidx;          // edge_index[0]
    const int* dst = eidx + E_;     // edge_index[1]

    // Workspace carve-up (all f32, 16B-aligned regions)
    float* ws      = (float*)d_ws;
    float* deg     = ws;                          // N   (becomes dinv)
    float* selfn   = deg   + N_;                  // N
    float* norm    = selfn + N_;                  // E
    float* hw      = norm  + E_;                  // N*D
    float* hA      = hw    + (size_t)N_ * D_;     // N*D
    float* hB      = hA    + (size_t)N_ * D_;     // N*D
    float* bnsum   = hB    + (size_t)N_ * D_;     // 64
    float* bnsq    = bnsum + 64;                  // 64
    float* bnscale = bnsq  + 64;                  // 64
    float* bnshift = bnscale + 64;                // 64
    float* meanacc = bnshift + 64;                // G*64
    float* mxbuf   = meanacc + (size_t)G_ * 64;   // G*64
    float* cnt     = mxbuf   + (size_t)G_ * 64;   // G

    const int nRowTiles = N_ / 16;                             // 3125 (exact)
    const int gemmBlocks = (nRowTiles + 3) / 4;                // 4 waves/block
    const int nb_N   = (N_ + 255) / 256;
    const int nb_E   = (E_ + 255) / 256;
    const int nb_N16 = (N_ * 16) / 256;                        // 3125
    const int nb_E16 = (int)(((long long)E_ * 16) / 256);      // 50000
    const int nb_N64 = (int)(((long long)N_ * 64) / 256);      // 12500

    // ---- GCN normalization coefficients
    k_deg_init<<<nb_N, 256, 0, stream>>>(deg, N_);
    k_deg_accum<<<nb_E, 256, 0, stream>>>(dst, eattr, deg, E_);
    k_dinv<<<nb_N, 256, 0, stream>>>(deg, selfn, N_);
    k_edge_norm<<<nb_E, 256, 0, stream>>>(src, dst, eattr, deg, norm, E_);

    // ---- Layer 0: x@W0 -> aggregate -> BN -> leaky   (hA)
    k_gemm_wmma<128><<<gemmBlocks, 128, 0, stream>>>(x, W0, hw, nRowTiles);
    k_agg_init<<<nb_N16, 256, 0, stream>>>(hw, selfn, b0, hA, N_);
    k_edge_scatter<<<nb_E16, 256, 0, stream>>>(hw, src, dst, norm, hA, E_);
    k_zero<<<1, 128, 0, stream>>>(bnsum, 128);                 // bnsum+bnsq contiguous
    k_bn_partial<<<256, 256, 0, stream>>>(hA, bnsum, bnsq, N_);
    k_bn_finalize<<<1, 64, 0, stream>>>(bnsum, bnsq, g0, be0, bnscale, bnshift, N_);
    k_bn_leaky<<<nb_N16, 256, 0, stream>>>(hA, bnscale, bnshift, N_ * 16);

    // ---- Layer 1: hA@W1 -> aggregate -> BN -> leaky  (hB)
    k_gemm_wmma<64><<<gemmBlocks, 128, 0, stream>>>(hA, W1, hw, nRowTiles);
    k_agg_init<<<nb_N16, 256, 0, stream>>>(hw, selfn, b1, hB, N_);
    k_edge_scatter<<<nb_E16, 256, 0, stream>>>(hw, src, dst, norm, hB, E_);
    k_zero<<<1, 128, 0, stream>>>(bnsum, 128);
    k_bn_partial<<<256, 256, 0, stream>>>(hB, bnsum, bnsq, N_);
    k_bn_finalize<<<1, 64, 0, stream>>>(bnsum, bnsq, g1, be1, bnscale, bnshift, N_);
    k_bn_leaky<<<nb_N16, 256, 0, stream>>>(hB, bnscale, bnshift, N_ * 16);

    // ---- Layer 2: hB@W2 -> aggregate -> leaky        (hA)
    k_gemm_wmma<64><<<gemmBlocks, 128, 0, stream>>>(hB, W2, hw, nRowTiles);
    k_agg_init<<<nb_N16, 256, 0, stream>>>(hw, selfn, b2, hA, N_);
    k_edge_scatter<<<nb_E16, 256, 0, stream>>>(hw, src, dst, norm, hA, E_);
    k_leaky<<<nb_N16, 256, 0, stream>>>(hA, N_ * 16);

    // ---- Layer 3: hA@W3 -> aggregate -> leaky        (hB)
    k_gemm_wmma<64><<<gemmBlocks, 128, 0, stream>>>(hA, W3, hw, nRowTiles);
    k_agg_init<<<nb_N16, 256, 0, stream>>>(hw, selfn, b3, hB, N_);
    k_edge_scatter<<<nb_E16, 256, 0, stream>>>(hw, src, dst, norm, hB, E_);
    k_leaky<<<nb_N16, 256, 0, stream>>>(hB, N_ * 16);

    // ---- Pooling + readout
    k_pool_init<<<(G_ * 64) / 256, 256, 0, stream>>>(meanacc, mxbuf, cnt);
    k_pool_scatter<<<nb_N64, 256, 0, stream>>>(hB, batch, meanacc, mxbuf, cnt, N_);
    k_readout<<<(G_ + 255) / 256, 256, 0, stream>>>(meanacc, mxbuf, cnt, Wout, bout, out);
}
